// FullAttention_146028888351
// MI455X (gfx1250) — compile-verified, matrix operand order checked by reading, not run on hardware
//
#include <hip/hip_runtime.h>
#include <math.h>

typedef __attribute__((ext_vector_type(16))) _Float16 v16h;
typedef __attribute__((ext_vector_type(8)))  _Float16 v8h;
typedef __attribute__((ext_vector_type(4)))  _Float16 v4h;
typedef __attribute__((ext_vector_type(2)))  __fp16   v2hf;   // return type of cvt_pkrtz
typedef __attribute__((ext_vector_type(8)))  float    v8f;

#define LOG2E 1.44269504088896f

// Problem constants from the reference
constexpr int Bc = 4, Lc = 2048, Sc = 2048, Hc = 8, Ec = 64, Dc = 64;
constexpr int QT   = 128;       // query rows per workgroup (8 waves x 16)
constexpr int NW   = 8;         // waves per block
constexpr int KT   = 64;        // kv columns per tile
constexpr int LSTR = 72;        // LDS row stride in halves (144B: 16B-aligned, conflict-free)

__global__ __launch_bounds__(256)
void fa_fwd_causal(const float* __restrict__ Q, const float* __restrict__ K,
                   const float* __restrict__ V, float* __restrict__ O)
{
    __shared__ _Float16 ldsK[KT * LSTR];          // [s][e] f16
    __shared__ _Float16 ldsV[Dc * LSTR];          // [d][s] f16 (transposed)
    __shared__ _Float16 ldsP[NW * 16 * LSTR];     // per-wave P scratch [m][s]

    const int tid  = threadIdx.x;
    const int lane = tid & 31;
    const int wv   = tid >> 5;
    const int lr   = lane & 15;                   // row/col index within 16
    const int hi   = (lane >> 4) & 1;             // which 16-lane half
    const int sel  = hi ? 8 : 0;                  // K-chunk select for A/B fragments

    const int qt = blockIdx.x;                    // 128-row query tile index (0..15)
    const int h  = blockIdx.y;
    const int b  = blockIdx.z;
    const int waveRow0 = qt * QT + wv * 16;

    const size_t strideRow = (size_t)Hc * Ec;     // 512 floats between consecutive l/s
    const float* Qb = Q + ((size_t)b * Lc * Hc + h) * Ec;
    const float* Kb = K + ((size_t)b * Sc * Hc + h) * Ec;
    const float* Vb = V + ((size_t)b * Sc * Hc + h) * Dc;
    float*       Ob = O + ((size_t)b * Lc * Hc + h) * Dc;

    // ---- Q A-fragments, pre-scaled by (1/sqrt(E)) * log2(e) so scores are ready
    //      for exp2-domain softmax with no per-element scaling in the loop ----
    const float qscl = 0.125f * LOG2E;
    v16h aq[2];
    {
        const float* qrow = Qb + (size_t)(waveRow0 + lr) * strideRow;
        #pragma unroll
        for (int c = 0; c < 2; ++c) {
            const int k0 = 32 * c;
            #pragma unroll
            for (int i = 0; i < 8; ++i) {
                aq[c][i]     = (_Float16)(qrow[k0 + sel + i] * qscl);
                aq[c][8 + i] = (_Float16)(qrow[k0 + 16 + sel + i] * qscl);
            }
        }
    }

    v8f o[4];
    #pragma unroll
    for (int t = 0; t < 4; ++t) o[t] = (v8f){};
    float m[8], l[8];
    #pragma unroll
    for (int r = 0; r < 8; ++r) { m[r] = -INFINITY; l[r] = 0.f; }

    // ---- vectorized cooperative staging of one 64-col KV tile ----
    auto stage = [&](int kv0) {
        // K tile, row-major [s][e]: 1024 float4 / 256 threads = 4 per thread
        #pragma unroll
        for (int k = 0; k < 4; ++k) {
            const int idx = tid + 256 * k;
            const int s = idx >> 4, e4 = (idx & 15) * 4;
            const float4 f = *(const float4*)(Kb + (size_t)(kv0 + s) * strideRow + e4);
            const v2hf lo = __builtin_amdgcn_cvt_pkrtz(f.x, f.y);
            const v2hf hh = __builtin_amdgcn_cvt_pkrtz(f.z, f.w);
            v4h h4;
            h4[0] = (_Float16)lo[0]; h4[1] = (_Float16)lo[1];
            h4[2] = (_Float16)hh[0]; h4[3] = (_Float16)hh[1];
            *(v4h*)(&ldsK[s * LSTR + e4]) = h4;
        }
        // V tile, transposed [d][s]: lane-contiguous s keeps LDS b16 scatter conflict-light
        #pragma unroll
        for (int k = 0; k < 4; ++k) {
            const int idx = tid + 256 * k;
            const int s = idx & 63, d = (idx >> 6) * 4;
            const float4 f = *(const float4*)(Vb + (size_t)(kv0 + s) * strideRow + d);
            ldsV[(d + 0) * LSTR + s] = (_Float16)f.x;
            ldsV[(d + 1) * LSTR + s] = (_Float16)f.y;
            ldsV[(d + 2) * LSTR + s] = (_Float16)f.z;
            ldsV[(d + 3) * LSTR + s] = (_Float16)f.w;
        }
    };

    // ---- one KV tile of flash attention (masked: apply causal cndmask) ----
    auto tile = [&](int kv0, bool masked) {
        // S = Q * K^T, 4 output tiles of 16x16, already in exp2 domain
        v8f acc[4];
        #pragma unroll
        for (int t = 0; t < 4; ++t) {
            acc[t] = (v8f){};
            #pragma unroll
            for (int c = 0; c < 2; ++c) {
                const _Float16* bp = &ldsK[(t * 16 + lr) * LSTR + 32 * c + sel];
                const v8h blo = *(const v8h*)bp;
                const v8h bhi = *(const v8h*)(bp + 16);
                v16h bfrag;
                #pragma unroll
                for (int i = 0; i < 8; ++i) { bfrag[i] = blo[i]; bfrag[8 + i] = bhi[i]; }
                acc[t] = __builtin_amdgcn_wmma_f32_16x16x32_f16(
                    false, aq[c], false, bfrag, (short)0, acc[t], false, false);
            }
        }

        float sv[4][8];
        const int myRowBase = waveRow0 + hi * 8;
        #pragma unroll
        for (int t = 0; t < 4; ++t) {
            const int gc = kv0 + t * 16 + lr;
            #pragma unroll
            for (int r = 0; r < 8; ++r) {
                float x = acc[t][r];
                if (masked && gc > myRowBase + r) x = -INFINITY;
                sv[t][r] = x;
            }
        }

        // online softmax (per-row, rows live across the 16-lane halves)
        #pragma unroll
        for (int r = 0; r < 8; ++r) {
            float rm = sv[0][r];
            #pragma unroll
            for (int t = 1; t < 4; ++t) rm = fmaxf(rm, sv[t][r]);
            #pragma unroll
            for (int d = 1; d < 16; d <<= 1)
                rm = fmaxf(rm, __shfl_xor(rm, d, 32));
            const float mnew = fmaxf(m[r], rm);
            const float fac  = exp2f(m[r] - mnew);
            m[r] = mnew;

            float rs = 0.f;
            #pragma unroll
            for (int t = 0; t < 4; ++t) {
                const float p = exp2f(sv[t][r] - mnew);
                sv[t][r] = p;
                rs += p;
            }
            #pragma unroll
            for (int d = 1; d < 16; d <<= 1)
                rs += __shfl_xor(rs, d, 32);
            l[r] = l[r] * fac + rs;
            #pragma unroll
            for (int t = 0; t < 4; ++t) o[t][r] *= fac;
        }

        // C-layout -> A-layout for P via wave-private LDS scratch.
        // Same-wave LDS ops are hardware in-order (DScnt pipe) => no barrier needed.
        _Float16* pbase = &ldsP[wv * 16 * LSTR];
        #pragma unroll
        for (int t = 0; t < 4; ++t) {
            #pragma unroll
            for (int r = 0; r < 8; ++r)
                pbase[(hi * 8 + r) * LSTR + t * 16 + lr] = (_Float16)sv[t][r];
        }

        // O += P * V
        #pragma unroll
        for (int c = 0; c < 2; ++c) {
            const int k0 = 32 * c;
            const _Float16* pr = &pbase[lr * LSTR + k0 + sel];
            const v8h plo = *(const v8h*)pr;
            const v8h phi = *(const v8h*)(pr + 16);
            v16h pa;
            #pragma unroll
            for (int i = 0; i < 8; ++i) { pa[i] = plo[i]; pa[8 + i] = phi[i]; }
            #pragma unroll
            for (int t = 0; t < 4; ++t) {
                const _Float16* vp = &ldsV[(t * 16 + lr) * LSTR + k0 + sel];
                const v8h vlo = *(const v8h*)vp;
                const v8h vhi = *(const v8h*)(vp + 16);
                v16h vb;
                #pragma unroll
                for (int i = 0; i < 8; ++i) { vb[i] = vlo[i]; vb[8 + i] = vhi[i]; }
                o[t] = __builtin_amdgcn_wmma_f32_16x16x32_f16(
                    false, pa, false, vb, (short)0, o[t], false, false);
            }
        }
    };

    // ---- causal tiling: tiles 0..2qt-1 need no mask; last two tiles are masked ----
    const int nFull = 2 * qt;
    for (int j = 0; j < nFull; ++j) {
        __syncthreads();
        stage(j * KT);
        __syncthreads();
        tile(j * KT, false);
    }
    #pragma unroll
    for (int e = 0; e < 2; ++e) {
        const int j = nFull + e;
        __syncthreads();
        stage(j * KT);
        __syncthreads();
        tile(j * KT, true);
    }

    // ---- normalize and store O [B,L,H,D] fp32 ----
    #pragma unroll
    for (int r = 0; r < 8; ++r) {
        const float inv = 1.0f / l[r];
        float* orow = Ob + (size_t)(waveRow0 + hi * 8 + r) * strideRow;
        #pragma unroll
        for (int t = 0; t < 4; ++t)
            orow[t * 16 + lr] = o[t][r] * inv;
    }
}

extern "C" void kernel_launch(void* const* d_in, const int* in_sizes, int n_in,
                              void* d_out, int out_size, void* d_ws, size_t ws_size,
                              hipStream_t stream) {
    (void)in_sizes; (void)n_in; (void)out_size; (void)d_ws; (void)ws_size;
    const float* Q = (const float*)d_in[0];
    const float* K = (const float*)d_in[1];
    const float* V = (const float*)d_in[2];
    // d_in[3] = attn_mask (triangular causal) — implemented analytically in-kernel
    float* O = (float*)d_out;

    dim3 grid(Lc / QT, Hc, Bc);   // (16, 8, 4)
    dim3 block(256);              // 8 waves (wave32)
    fa_fwd_causal<<<grid, block, 0, stream>>>(Q, K, V, O);
}
